// MultiheadAttention_28991029248378
// MI455X (gfx1250) — matrix operand order repaired and checked by reading, not compiled, not run on hardware
//
#include <hip/hip_runtime.h>

// ---------------------------------------------------------------------------
// Types for CDNA5 WMMA bf16 path
// ---------------------------------------------------------------------------
typedef __attribute__((ext_vector_type(16))) __bf16          v16bf;
typedef __attribute__((ext_vector_type(8)))  float           v8f;
typedef __attribute__((ext_vector_type(8)))  unsigned short  v8u;
typedef __attribute__((ext_vector_type(4)))  float           f32x4;
typedef __attribute__((ext_vector_type(4)))  unsigned short  u16x4;

union BF16x16 {
    unsigned short us[16];
    v16bf b;
};

#define WMMA_BF16(A, B, C) \
    __builtin_amdgcn_wmma_f32_16x16x32_bf16(false, (A), false, (B), (short)0, (C), false, false)

// ---------------------------------------------------------------------------
// Scalar bf16 helpers (round-to-nearest-even)
// ---------------------------------------------------------------------------
__device__ __forceinline__ unsigned short f2bf(float f) {
    union { float f; unsigned u; } c; c.f = f;
    unsigned r = c.u + 0x7FFFu + ((c.u >> 16) & 1u);
    return (unsigned short)(r >> 16);
}
__device__ __forceinline__ float bf2f(unsigned short h) {
    union { unsigned u; float f; } c; c.u = ((unsigned)h) << 16;
    return c.f;
}

// A-operand (16x32, row per lane): elems 0..7 = K grp*8+0..7, elems 8..15 = K grp*8+16..23
__device__ __forceinline__ v16bf load_mat_a(const unsigned short* rowp, int grp) {
    BF16x16 t;
    ((v8u*)&t)[0] = *(const v8u*)(rowp + grp * 8);
    ((v8u*)&t)[1] = *(const v8u*)(rowp + grp * 8 + 16);
    return t.b;
}
// B-operand (32x16, col per lane = B^T row): elems 0..7 = K grp*8+0..7, elems 8..15 = K grp*8+16..23
__device__ __forceinline__ v16bf load_mat_b(const unsigned short* rowp, int grp) {
    BF16x16 t;
    ((v8u*)&t)[0] = *(const v8u*)(rowp + grp * 8);
    ((v8u*)&t)[1] = *(const v8u*)(rowp + grp * 8 + 16);
    return t.b;
}

// ---------------------------------------------------------------------------
// Problem constants (reference: B=4, N=2048, D=E=512, H=8, hd=64, TOP_K=32)
// ---------------------------------------------------------------------------
#define BB   4
#define NN   2048
#define DD   512
#define EE   512
#define HH   8
#define HD   64
#define TOPK 32

// ---------------------------------------------------------------------------
// f32 -> bf16 conversion, 4 elements / thread
// ---------------------------------------------------------------------------
__global__ void cvt_f32_bf16_x4(const f32x4* __restrict__ in,
                                u16x4* __restrict__ out, int n4) {
    int i = blockIdx.x * 256 + threadIdx.x;
    if (i < n4) {
        f32x4 v = in[i];
        u16x4 o;
        o[0] = f2bf(v[0]); o[1] = f2bf(v[1]); o[2] = f2bf(v[2]); o[3] = f2bf(v[3]);
        out[i] = o;
    }
}

// ---------------------------------------------------------------------------
// GEMM1: qkv[m, c] = sum_k xb[m,k] * Wqkv[c,k], scattered to Q/K/V [B,H,N,64]
// One wave per 16(M) x 64(N) tile: A operand reused by 4 accumulators.
// grid = (1536/64, 8192/16), block = 32.  Q gets 1/sqrt(hd)=0.125 folded in.
// ---------------------------------------------------------------------------
__global__ __launch_bounds__(32) void gemm_qkv(
    const unsigned short* __restrict__ xb,
    const unsigned short* __restrict__ wb,
    unsigned short* __restrict__ Qb,
    unsigned short* __restrict__ Kb,
    unsigned short* __restrict__ Vb) {
    const int lane = threadIdx.x;
    const int grp  = lane >> 4;
    const int l16  = lane & 15;
    const int row     = blockIdx.y * 16 + l16;   // M index for A load
    const int colbase = blockIdx.x * 64;
    const unsigned short* arow = xb + (size_t)row * DD;
    const unsigned short* b0 = wb + (size_t)(colbase +  0 + l16) * DD;
    const unsigned short* b1 = wb + (size_t)(colbase + 16 + l16) * DD;
    const unsigned short* b2 = wb + (size_t)(colbase + 32 + l16) * DD;
    const unsigned short* b3 = wb + (size_t)(colbase + 48 + l16) * DD;

    v8f acc0 = {}, acc1 = {}, acc2 = {}, acc3 = {};
#pragma unroll
    for (int k0 = 0; k0 < DD; k0 += 32) {
        v16bf a = load_mat_a(arow + k0, grp);
        acc0 = WMMA_BF16(a, load_mat_b(b0 + k0, grp), acc0);
        acc1 = WMMA_BF16(a, load_mat_b(b1 + k0, grp), acc1);
        acc2 = WMMA_BF16(a, load_mat_b(b2 + k0, grp), acc2);
        acc3 = WMMA_BF16(a, load_mat_b(b3 + k0, grp), acc3);
    }

    auto store_tile = [&](const v8f& acc, int col) {
        const int h   = col / (3 * HD);
        const int rem = col % (3 * HD);
        const int t   = rem / HD;
        const int dd  = rem % HD;
        const float scale = (t == 0) ? 0.125f : 1.0f;
        unsigned short* outp = (t == 0) ? Qb : ((t == 1) ? Kb : Vb);
#pragma unroll
        for (int r = 0; r < 8; ++r) {
            const int m    = blockIdx.y * 16 + grp * 8 + r;  // C/D: lanes16-31 hold M=8..15
            const int bidx = m >> 11;
            const int nrow = m & (NN - 1);
            const size_t o = (((size_t)bidx * HH + h) * NN + nrow) * HD + dd;
            outp[o] = f2bf(acc[r] * scale);
        }
    };
    store_tile(acc0, colbase +  0 + l16);
    store_tile(acc1, colbase + 16 + l16);
    store_tile(acc2, colbase + 32 + l16);
    store_tile(acc3, colbase + 48 + l16);
}

// ---------------------------------------------------------------------------
// Attention with top-k: one block per (b, h, 16-query tile). 256 threads.
// Q tile async-staged Global->LDS (ASYNCcnt); score slab S[16][2048] in
// dynamic LDS (128 KB of the 320 KB WGP LDS).
// ---------------------------------------------------------------------------
__global__ __launch_bounds__(256) void attn_topk(
    const unsigned short* __restrict__ Qb,
    const unsigned short* __restrict__ Kb,
    const unsigned short* __restrict__ Vb,
    unsigned short* __restrict__ Hb) {
    extern __shared__ float S[];                 // [16][NN]
    __shared__ unsigned short Qtile[16 * HD];    // 2 KB staged via async DMA
    __shared__ int   selIdx[16 * TOPK];
    __shared__ float selW[16 * TOPK];

    const int tid  = threadIdx.x;
    const int wave = tid >> 5;
    const int lane = tid & 31;
    const int grp  = lane >> 4;
    const int l16  = lane & 15;
    const int qt   = blockIdx.x;                 // 0..127
    const int h    = blockIdx.y;
    const int b    = blockIdx.z;
    const size_t bh = ((size_t)b * HH + h) * NN;

    // ---- Phase 0: async-stage Q tile (16x64 bf16 = 2 KB = 128 lanes x 16B) ----
    if (tid < 128) {
        const unsigned short* gsrc = Qb + (bh + qt * 16) * HD + tid * 8;
        unsigned lds_off = (unsigned)(uintptr_t)(&Qtile[tid * 8]);  // addr[31:0] = LDS offset
        asm volatile("global_load_async_to_lds_b128 %0, %1, off"
                     :: "v"(lds_off), "v"((unsigned long long)(uintptr_t)gsrc)
                     : "memory");
    }
    asm volatile("s_wait_asynccnt 0x0" ::: "memory");
    __syncthreads();

    // ---- Phase 1: S[q, k] = Q . K  (scale already folded into Q) ----
    const unsigned short* qrow = &Qtile[l16 * HD];
    const v16bf aq0 = load_mat_a(qrow,      grp);   // K-dim 0..31  (ds_load_b128)
    const v16bf aq1 = load_mat_a(qrow + 32, grp);   // K-dim 32..63

#pragma unroll
    for (int t = 0; t < 16; ++t) {
        const int ct = wave * 256 + t * 16;      // this wave's key-column tile
        const unsigned short* kptr = Kb + (bh + ct + l16) * HD;
        v16bf bk0 = load_mat_b(kptr,      grp);
        v16bf bk1 = load_mat_b(kptr + 32, grp);
        v8f acc = {};
        acc = WMMA_BF16(aq0, bk0, acc);
        acc = WMMA_BF16(aq1, bk1, acc);
        const int colS = ct + l16;
#pragma unroll
        for (int r = 0; r < 8; ++r)
            S[(grp * 8 + r) * NN + colS] = acc[r];
    }
    __syncthreads();

    // ---- Phase 2: per-row softmax stats + iterative top-32 (wave owns 2 rows) ----
    for (int rp = 0; rp < 2; ++rp) {
        const int row = wave * 2 + rp;
        float* Srow = S + row * NN;

        float m = -3.402823466e38f;
        for (int i = lane; i < NN; i += 32) m = fmaxf(m, Srow[i]);
        for (int off = 16; off; off >>= 1) m = fmaxf(m, __shfl_xor(m, off, 32));

        float d = 0.0f;
        for (int i = lane; i < NN; i += 32) d += __expf(Srow[i] - m);
        for (int off = 16; off; off >>= 1) d += __shfl_xor(d, off, 32);
        const float inv = 1.0f / d;              // denom over FULL row (ref: softmax then mask)

        for (int s = 0; s < TOPK; ++s) {
            float bv = -3.402823466e38f; int bi = 0x7FFFFFFF;
            for (int i = lane; i < NN; i += 32) {
                float v = Srow[i];
                if (v > bv) { bv = v; bi = i; }
            }
            for (int off = 16; off; off >>= 1) {
                float ov = __shfl_xor(bv, off, 32);
                int   oi = __shfl_xor(bi, off, 32);
                if (ov > bv || (ov == bv && oi < bi)) { bv = ov; bi = oi; }
            }
            if (lane == (bi & 31)) {             // owning lane records + knocks out
                Srow[bi] = -3.402823466e38f;
                selIdx[row * TOPK + s] = bi;
                selW[row * TOPK + s]   = __expf(bv - m) * inv;
            }
        }
    }
    __syncthreads();

    // ---- Phase 3: sparse A@V gather, hd=64 -> 2 dims per lane ----
    for (int rp = 0; rp < 2; ++rp) {
        const int row = wave * 2 + rp;
        float a0 = 0.0f, a1 = 0.0f;
#pragma unroll
        for (int s = 0; s < TOPK; ++s) {
            const float w  = selW[row * TOPK + s];
            const int   ki = selIdx[row * TOPK + s];
            const unsigned short* vptr = Vb + (bh + ki) * HD;
            a0 += w * bf2f(vptr[lane]);
            a1 += w * bf2f(vptr[lane + 32]);
        }
        const int q = qt * 16 + row;
        unsigned short* op = Hb + (((size_t)b * NN + q) * HH + h) * HD;  // [B,N,E]
        op[lane]      = f2bf(a0);
        op[lane + 32] = f2bf(a1);
    }
}

// ---------------------------------------------------------------------------
// GEMM2: out[m, c] = sum_k Hb[m,k] * Wo[c,k]  (f32 output)
// One wave per 16x64 tile. grid = (512/64, 8192/16), block = 32.
// ---------------------------------------------------------------------------
__global__ __launch_bounds__(32) void gemm_out(
    const unsigned short* __restrict__ hb,
    const unsigned short* __restrict__ wob,
    float* __restrict__ out) {
    const int lane = threadIdx.x;
    const int grp  = lane >> 4;
    const int l16  = lane & 15;
    const int row     = blockIdx.y * 16 + l16;
    const int colbase = blockIdx.x * 64;
    const unsigned short* arow = hb + (size_t)row * EE;
    const unsigned short* b0 = wob + (size_t)(colbase +  0 + l16) * EE;
    const unsigned short* b1 = wob + (size_t)(colbase + 16 + l16) * EE;
    const unsigned short* b2 = wob + (size_t)(colbase + 32 + l16) * EE;
    const unsigned short* b3 = wob + (size_t)(colbase + 48 + l16) * EE;

    v8f acc0 = {}, acc1 = {}, acc2 = {}, acc3 = {};
#pragma unroll
    for (int k0 = 0; k0 < EE; k0 += 32) {
        v16bf a = load_mat_a(arow + k0, grp);
        acc0 = WMMA_BF16(a, load_mat_b(b0 + k0, grp), acc0);
        acc1 = WMMA_BF16(a, load_mat_b(b1 + k0, grp), acc1);
        acc2 = WMMA_BF16(a, load_mat_b(b2 + k0, grp), acc2);
        acc3 = WMMA_BF16(a, load_mat_b(b3 + k0, grp), acc3);
    }
    auto store_tile = [&](const v8f& acc, int col) {
#pragma unroll
        for (int r = 0; r < 8; ++r) {
            const int m = blockIdx.y * 16 + grp * 8 + r;
            out[(size_t)m * EE + col] = acc[r];
        }
    };
    store_tile(acc0, colbase +  0 + l16);
    store_tile(acc1, colbase + 16 + l16);
    store_tile(acc2, colbase + 32 + l16);
    store_tile(acc3, colbase + 48 + l16);
}

// ---------------------------------------------------------------------------
// Host launcher
// ---------------------------------------------------------------------------
extern "C" void kernel_launch(void* const* d_in, const int* in_sizes, int n_in,
                              void* d_out, int out_size, void* d_ws, size_t ws_size,
                              hipStream_t stream) {
    (void)in_sizes; (void)n_in; (void)out_size; (void)ws_size;
    const float* x    = (const float*)d_in[0];
    const float* Wqkv = (const float*)d_in[1];
    const float* Wo   = (const float*)d_in[2];
    float* out = (float*)d_out;

    size_t off = 0;
    auto carve = [&](size_t elems) -> unsigned short* {
        unsigned short* p = (unsigned short*)((char*)d_ws + off);
        off += ((elems * sizeof(unsigned short)) + 255) & ~(size_t)255;
        return p;
    };
    unsigned short* xb    = carve((size_t)BB * NN * DD);      //  8.4 MB
    unsigned short* wqkvb = carve((size_t)3 * EE * DD);       //  1.5 MB
    unsigned short* wob   = carve((size_t)EE * EE);           //  0.5 MB
    unsigned short* Qb    = carve((size_t)BB * HH * NN * HD); //  8.4 MB
    unsigned short* Kb    = carve((size_t)BB * HH * NN * HD);
    unsigned short* Vb    = carve((size_t)BB * HH * NN * HD);
    unsigned short* Hb    = carve((size_t)BB * NN * EE);      //  8.4 MB  (~44 MB total)

    const int nx = BB * NN * DD, nwq = 3 * EE * DD, nwo = EE * EE;
    cvt_f32_bf16_x4<<<nx / 1024,  256, 0, stream>>>((const f32x4*)x,    (u16x4*)xb,    nx / 4);
    cvt_f32_bf16_x4<<<nwq / 1024, 256, 0, stream>>>((const f32x4*)Wqkv, (u16x4*)wqkvb, nwq / 4);
    cvt_f32_bf16_x4<<<nwo / 1024, 256, 0, stream>>>((const f32x4*)Wo,   (u16x4*)wob,   nwo / 4);

    gemm_qkv<<<dim3(3 * EE / 64, BB * NN / 16), 32, 0, stream>>>(xb, wqkvb, Qb, Kb, Vb);

    attn_topk<<<dim3(NN / 16, HH, BB), 256, 16 * NN * sizeof(float), stream>>>(Qb, Kb, Vb, Hb);

    gemm_out<<<dim3(EE / 64, BB * NN / 16), 32, 0, stream>>>(Hb, wob, out);
}
